// MultiheadGQA_45947560132743
// MI455X (gfx1250) — matrix operand order repaired and checked by reading, not compiled, not run on hardware
//
#include <hip/hip_runtime.h>
#include <hip/hip_bf16.h>

typedef __attribute__((ext_vector_type(16))) __bf16 v16bf;
typedef __attribute__((ext_vector_type(8)))  float  v8f;
typedef __attribute__((ext_vector_type(4)))  unsigned int v4u;
typedef __attribute__((ext_vector_type(8)))  int v8i_;
typedef __attribute__((ext_vector_type(4)))  int v4i_;

union Frag16 { v16bf v; unsigned int u[8]; };

#define B_  2
#define N_  2048
#define E_  2048
#define HQ_ 16
#define HK_ 4
#define D_  128
#define G_  4
#define KVE_ 512

// TDM path: probe-verified 5-arg builtin exists on ROCm 7.2 (clang-22); the
// amdgpu-toolchain ships the TDM header and a 6-arg builtin instead — detect
// via __has_include and fall back to the register-staged copy there.
#if defined(__has_builtin)
#if __has_builtin(__builtin_amdgcn_tensor_load_to_lds) && \
    __has_builtin(__builtin_amdgcn_s_wait_tensorcnt) && \
    !__has_include(<hip/amd_detail/amd_gfx1250_TDM.h>)
#define USE_TDM 1
#endif
#endif
#ifndef USE_TDM
#define USE_TDM 0
#endif

__device__ __forceinline__ unsigned short f32_to_bf16_rne(float f) {
  unsigned int u = __float_as_uint(f);
  unsigned int r = u + 0x7FFFu + ((u >> 16) & 1u);
  return (unsigned short)(r >> 16);
}

__device__ __forceinline__ v8f wmma_bf16(Frag16 a, Frag16 b, v8f c) {
  return __builtin_amdgcn_wmma_f32_16x16x32_bf16(
      /*neg_a=*/false, a.v, /*neg_b=*/false, b.v,
      /*c_mod=*/(short)0, c, /*reuse_a=*/false, /*reuse_b=*/false);
}

#if USE_TDM
// 2-D tensor tile -> LDS via Tensor Data Mover. D# per CDNA5 ISA:
// group0: [1:0]count=1, [63:32]lds_addr, [120:64]global_addr, [127:126]type=2
// group1: [17:16]data_size(1=2B), [20]pad_enable, [24:22]pad_interval,
//         [31:25]pad_amount(0 => 1 DWORD), [79:48]tensor_dim0,
//         [111:80]tensor_dim1, [127:112]tile_dim0, [143:128]tile_dim1,
//         [207:160]tensor_dim0_stride.  (2-D: groups 2/3 zero)
__device__ __forceinline__ void tdm_load_2d(unsigned lds_off, const void* gaddr,
                                            unsigned tile_d0, unsigned tile_d1,
                                            unsigned stride_d0,
                                            unsigned pad_code) {
  unsigned long long ga = (unsigned long long)gaddr;
  unsigned td0 = stride_d0;        // tensor_dim0 >= tile_d0 (no OOB zeroing)
  unsigned td1 = tile_d1;
  v4u g0;
  g0.x = 1u;                                            // count=1
  g0.y = lds_off;                                       // LDS byte address
  g0.z = (unsigned)ga;                                  // global addr lo
  g0.w = (unsigned)((ga >> 32) & 0x01FFFFFFull) | 0x80000000u;  // type=2
  v8i_ g1;
  g1[0] = (int)((1u << 16) | (1u << 20) | (pad_code << 22));  // 2B, pad
  g1[1] = (int)((td0 & 0xFFFFu) << 16);
  g1[2] = (int)(((td0 >> 16) & 0xFFFFu) | ((td1 & 0xFFFFu) << 16));
  g1[3] = (int)(((td1 >> 16) & 0xFFFFu) | ((tile_d0 & 0xFFFFu) << 16));
  g1[4] = (int)(tile_d1 & 0xFFFFu);                     // tile_dim2 = 0
  g1[5] = (int)stride_d0;                               // stride[31:0]
  g1[6] = 0;
  g1[7] = 0;
  v4i_ z = {0, 0, 0, 0};
  __builtin_amdgcn_tensor_load_to_lds(g0, g1, z, z, 0);
}
__device__ __forceinline__ unsigned lds_offset_of(const void* p) {
  return (unsigned)(unsigned long long)p;  // low 32 bits of generic = LDS addr
}
#endif

// ---------------------------------------------------------------------------
// GEMM: C[M,Nd] = A[M,Kd] * W[Kd,Nd] + bias[Nd].
// 128x128 block tile, 256 threads (8 waves, each 32x64). K-step 32.
// Double-buffered LDS (one barrier per K-step); next tile's global loads are
// in flight during the current tile's WMMAs. LDS rows padded to 34 ushorts
// (17 banks) for conflict-free fragment reads.
// OUT_MODE: 0 = f32 row-major, 1 = bf16 row-major, 2 = bf16 KV-transposed
// (Vt[b][col][s], for the attention V operand).
// ---------------------------------------------------------------------------
template <bool A_BF16, int OUT_MODE>
__global__ void __launch_bounds__(256)
gemm_wmma_kernel(const void* __restrict__ Av, const float* __restrict__ W,
                 const float* __restrict__ bias, void* __restrict__ Cv,
                 int M, int Kd, int Nd) {
  __shared__ unsigned short lds_a[2][128][34];   // [buf][row m][k]
  __shared__ unsigned short lds_b[2][128][34];   // [buf][col n][k]
  const int tid  = threadIdx.x;
  const int lane = tid & 31;
  const int half = lane >> 4;
  const int l16  = lane & 15;
  const int wave = tid >> 5;
  const int wm0  = (wave >> 1) * 32;   // 4 wave-rows of 32
  const int wn0  = (wave & 1) * 64;    // 2 wave-cols of 64
  const int m0   = blockIdx.y * 128;
  const int n0   = blockIdx.x * 128;

  unsigned int aReg[8];
  unsigned int wReg[8];

  auto loadTile = [&](int k0) {
#pragma unroll
    for (int i = 0; i < 8; ++i) {              // A: 128 rows x 16 k-pairs
      int idx = tid + i * 256;
      int row = idx >> 4, kp = idx & 15;
      if (A_BF16) {
        const unsigned short* A = (const unsigned short*)Av;
        aReg[i] = *(const unsigned int*)(A + (size_t)(m0 + row) * Kd + k0 + kp * 2);
      } else {
        const float* A = (const float*)Av;
        float2 f = *(const float2*)(A + (size_t)(m0 + row) * Kd + k0 + kp * 2);
        aReg[i] = (unsigned int)f32_to_bf16_rne(f.x) |
                  ((unsigned int)f32_to_bf16_rne(f.y) << 16);
      }
    }
#pragma unroll
    for (int i = 0; i < 8; ++i) {              // W: 16 k-pairs x 128 cols
      int idx = tid + i * 256;
      int kp = idx >> 7, n = idx & 127;
      float w0 = W[(size_t)(k0 + 2 * kp) * Nd + n0 + n];
      float w1 = W[(size_t)(k0 + 2 * kp + 1) * Nd + n0 + n];
      wReg[i] = (unsigned int)f32_to_bf16_rne(w0) |
                ((unsigned int)f32_to_bf16_rne(w1) << 16);
    }
  };
  auto storeTile = [&](int buf) {
#pragma unroll
    for (int i = 0; i < 8; ++i) {
      int idx = tid + i * 256;
      int row = idx >> 4, kp = idx & 15;
      *(unsigned int*)&lds_a[buf][row][kp * 2] = aReg[i];
    }
#pragma unroll
    for (int i = 0; i < 8; ++i) {
      int idx = tid + i * 256;
      int kp = idx >> 7, n = idx & 127;
      *(unsigned int*)&lds_b[buf][n][kp * 2] = wReg[i];
    }
  };

  v8f acc[2][4] = {};

  loadTile(0);
  storeTile(0);
  __syncthreads();

  for (int k0 = 0; k0 < Kd; k0 += 32) {
    const int  cur = (k0 >> 5) & 1;
    const bool hasNext = (k0 + 32) < Kd;
    if (hasNext) loadTile(k0 + 32);   // in flight during WMMAs below

    Frag16 af[2], bf[4];
#pragma unroll
    for (int t = 0; t < 2; ++t)
#pragma unroll
      for (int i = 0; i < 8; ++i) {
        int ka = ((i >> 2) << 4) + (half << 3) + ((i & 3) << 1);
        af[t].u[i] = *(const unsigned int*)&lds_a[cur][wm0 + 16 * t + l16][ka];
      }
#pragma unroll
    for (int t = 0; t < 4; ++t)
#pragma unroll
      for (int i = 0; i < 8; ++i) {
        int kb = ((i >> 2) << 4) + (half << 3) + ((i & 3) << 1);
        bf[t].u[i] = *(const unsigned int*)&lds_b[cur][wn0 + 16 * t + l16][kb];
      }
#pragma unroll
    for (int ta = 0; ta < 2; ++ta)
#pragma unroll
      for (int tb = 0; tb < 4; ++tb)
        acc[ta][tb] = wmma_bf16(af[ta], bf[tb], acc[ta][tb]);

    if (hasNext) {
      storeTile(cur ^ 1);   // other buffer: no conflict with cur readers
      __syncthreads();
    }
  }

  // epilogue: C layout -> m = r + 8*half, n = l16
#pragma unroll
  for (int ta = 0; ta < 2; ++ta)
#pragma unroll
    for (int tb = 0; tb < 4; ++tb)
#pragma unroll
      for (int r = 0; r < 8; ++r) {
        int mrow = m0 + wm0 + 16 * ta + r + half * 8;
        int ncol = n0 + wn0 + 16 * tb + l16;
        float val = acc[ta][tb][r] + bias[ncol];
        if (OUT_MODE == 0) {
          ((float*)Cv)[(size_t)mrow * Nd + ncol] = val;
        } else if (OUT_MODE == 1) {
          ((unsigned short*)Cv)[(size_t)mrow * Nd + ncol] = f32_to_bf16_rne(val);
        } else {  // KV-transposed: Vt[b][col][s]
          int bb = mrow >> 11;              // mrow / N_
          int ss = mrow & (N_ - 1);
          ((unsigned short*)Cv)[((size_t)bb * KVE_ + ncol) * N_ + ss] =
              f32_to_bf16_rne(val);
        }
      }
}

// ---------------------------------------------------------------------------
// Grouped-query flash attention.
// Block = (batch, kv-head, 16-row q tile); its 4 waves are the 4 query heads
// of that KV group, so K/V chunks are shared through LDS. V comes from the
// pre-transposed Vt[b][col][s] so both TDM and fallback staging are pure row
// copies. Double-buffered (one barrier per 32-wide s chunk). On ROCm 7.2 the
// staging uses TENSOR_LOAD_TO_LDS with hardware LDS padding (260B / 68B row
// strides = the 17-bank anti-conflict layout).
// ---------------------------------------------------------------------------
__global__ void __launch_bounds__(128)
gqa_attn_kernel(const unsigned short* __restrict__ Qb,
                const unsigned short* __restrict__ Kb,
                const unsigned short* __restrict__ Vt,
                float* __restrict__ X) {
  __shared__ unsigned short ldsK[2][32][130];    // [buf][s][d]  260B rows
  __shared__ unsigned short ldsV[2][128][34];    // [buf][d][s]   68B rows
  __shared__ unsigned short ldsP[4][16][34];     // per-wave P transpose slab
  const int tid = threadIdx.x;
  const int lane = tid & 31, half = lane >> 4, l16 = lane & 15;
  const int wave = tid >> 5;                     // = group index g
  const int nt = N_ / 16;
  const int rowTile = blockIdx.x % nt;
  const int bh = blockIdx.x / nt;                // 0..B*HK-1
  const int b  = bh / HK_;
  const int hk = bh % HK_;
  const int hq = hk * G_ + wave;
  const int rowbase = rowTile * 16;
  const int qcol = hq * D_;
  const int kcol = hk * D_;

  // Q fragments: 16 rows x 128 (4 chunks of K=32), loaded once
  Frag16 qf[4];
#pragma unroll
  for (int kc = 0; kc < 4; ++kc)
#pragma unroll
    for (int i = 0; i < 8; ++i) {
      int ka = ((i >> 2) << 4) + (half << 3) + ((i & 3) << 1);
      int row = rowbase + l16;
      qf[kc].u[i] = *(const unsigned int*)(Qb + (size_t)(b * N_ + row) * E_ +
                                           qcol + kc * 32 + ka);
    }

#if USE_TDM
  auto issueKV = [&](int sc, int buf) {
    // K chunk: 32 rows x 128 elems, stride KVE; pad 1 DW per 64 DW (256B row)
    tdm_load_2d(lds_offset_of(&ldsK[buf][0][0]),
                Kb + (size_t)(b * N_ + sc) * KVE_ + kcol,
                /*tile_d0=*/D_, /*tile_d1=*/32, /*stride=*/KVE_,
                /*pad_code=*/5);
    // Vt chunk: 128 rows x 32 elems, stride N; pad 1 DW per 16 DW (64B row)
    tdm_load_2d(lds_offset_of(&ldsV[buf][0][0]),
                Vt + ((size_t)b * KVE_ + kcol) * N_ + sc,
                /*tile_d0=*/32, /*tile_d1=*/D_, /*stride=*/N_,
                /*pad_code=*/3);
  };
#else
  uint4 kReg[4], vReg[4];
  auto loadKV = [&](int sc) {
#pragma unroll
    for (int j = 0; j < 4; ++j) {              // K: 32 rows x 16 uint4
      int idx = tid + j * 128;
      int row = idx >> 4, q = idx & 15;
      kReg[j] = *(const uint4*)(Kb + (size_t)(b * N_ + sc + row) * KVE_ +
                                kcol + q * 8);
    }
#pragma unroll
    for (int j = 0; j < 4; ++j) {              // Vt: 128 rows x 4 uint4
      int idx = tid + j * 128;
      int d = idx >> 2, q = idx & 3;
      vReg[j] = *(const uint4*)(Vt + ((size_t)b * KVE_ + kcol + d) * N_ +
                                sc + q * 8);
    }
  };
  auto storeKV = [&](int buf) {
#pragma unroll
    for (int j = 0; j < 4; ++j) {
      int idx = tid + j * 128;
      int row = idx >> 4, q = idx & 15;
      *(unsigned int*)&ldsK[buf][row][q * 8 + 0] = kReg[j].x;
      *(unsigned int*)&ldsK[buf][row][q * 8 + 2] = kReg[j].y;
      *(unsigned int*)&ldsK[buf][row][q * 8 + 4] = kReg[j].z;
      *(unsigned int*)&ldsK[buf][row][q * 8 + 6] = kReg[j].w;
    }
#pragma unroll
    for (int j = 0; j < 4; ++j) {
      int idx = tid + j * 128;
      int d = idx >> 2, q = idx & 3;
      *(unsigned int*)&ldsV[buf][d][q * 8 + 0] = vReg[j].x;
      *(unsigned int*)&ldsV[buf][d][q * 8 + 2] = vReg[j].y;
      *(unsigned int*)&ldsV[buf][d][q * 8 + 4] = vReg[j].z;
      *(unsigned int*)&ldsV[buf][d][q * 8 + 6] = vReg[j].w;
    }
  };
#endif

  v8f acc[8] = {};
  float mrun[8], lrun[8];
#pragma unroll
  for (int r = 0; r < 8; ++r) { mrun[r] = -__builtin_inff(); lrun[r] = 0.f; }
  const float scale = 0.08838834764831845f;  // 1/sqrt(128)

#if USE_TDM
  if (wave == 0) {
    issueKV(0, 0);
    __builtin_amdgcn_s_wait_tensorcnt((short)0);
  }
  __syncthreads();
#else
  loadKV(0);
  storeKV(0);
  __syncthreads();
#endif

  for (int sc = 0; sc < N_; sc += 32) {
    const int  cur = (sc >> 5) & 1;
    const bool hasNext = (sc + 32) < N_;
    if (hasNext) {
#if USE_TDM
      if (wave == 0) issueKV(sc + 32, cur ^ 1);   // DMA overlaps compute
#else
      loadKV(sc + 32);                            // VMEM overlaps compute
#endif
    }

    // sim tiles from LDS K
    v8f c[2] = {};
#pragma unroll
    for (int t = 0; t < 2; ++t)
#pragma unroll
      for (int kc = 0; kc < 4; ++kc) {
        Frag16 kf;
#pragma unroll
        for (int i = 0; i < 8; ++i) {
          int kb = ((i >> 2) << 4) + (half << 3) + ((i & 3) << 1);
          kf.u[i] = *(const unsigned int*)&ldsK[cur][16 * t + l16][kc * 32 + kb];
        }
        c[t] = wmma_bf16(qf[kc], kf, c[t]);
      }

    // online softmax (row r + 8*half, col l16)
#pragma unroll
    for (int r = 0; r < 8; ++r) {
      c[0][r] *= scale; c[1][r] *= scale;
      float mx = fmaxf(c[0][r], c[1][r]);
      mx = fmaxf(mx, __shfl_xor(mx, 1, 32));
      mx = fmaxf(mx, __shfl_xor(mx, 2, 32));
      mx = fmaxf(mx, __shfl_xor(mx, 4, 32));
      mx = fmaxf(mx, __shfl_xor(mx, 8, 32));
      float mnew = fmaxf(mrun[r], mx);
      float corr = __expf(mrun[r] - mnew);
      float p0 = __expf(c[0][r] - mnew);
      float p1 = __expf(c[1][r] - mnew);
      c[0][r] = p0; c[1][r] = p1;
      float rs = p0 + p1;
      rs += __shfl_xor(rs, 1, 32);
      rs += __shfl_xor(rs, 2, 32);
      rs += __shfl_xor(rs, 4, 32);
      rs += __shfl_xor(rs, 8, 32);
      lrun[r] = lrun[r] * corr + rs;
      mrun[r] = mnew;
#pragma unroll
      for (int d = 0; d < 8; ++d) acc[d][r] *= corr;
    }

    // P: C-layout -> A-layout transpose through per-wave LDS slab
#pragma unroll
    for (int t = 0; t < 2; ++t)
#pragma unroll
      for (int r = 0; r < 8; ++r)
        ldsP[wave][r + half * 8][16 * t + l16] = f32_to_bf16_rne(c[t][r]);
    Frag16 pf;
#pragma unroll
    for (int i = 0; i < 8; ++i) {
      int ka = ((i >> 2) << 4) + (half << 3) + ((i & 3) << 1);
      pf.u[i] = *(const unsigned int*)&ldsP[wave][l16][ka];
    }

    // out += P(16x32) * V(32x128), V fragments from transposed LDS
#pragma unroll
    for (int d = 0; d < 8; ++d) {
      Frag16 vf;
#pragma unroll
      for (int i = 0; i < 8; ++i) {
        int kb = ((i >> 2) << 4) + (half << 3) + ((i & 3) << 1);
        vf.u[i] = *(const unsigned int*)&ldsV[cur][d * 16 + l16][kb];
      }
      acc[d] = wmma_bf16(pf, vf, acc[d]);
    }

    if (hasNext) {
#if USE_TDM
      if (wave == 0) __builtin_amdgcn_s_wait_tensorcnt((short)0);
#else
      storeKV(cur ^ 1);   // other buffer: no conflict with cur readers
#endif
      __syncthreads();
    }
  }

  // normalize and write f32 attention output
#pragma unroll
  for (int r = 0; r < 8; ++r) {
    float inv = 1.0f / lrun[r];
    int row = rowbase + r + half * 8;
#pragma unroll
    for (int d = 0; d < 8; ++d)
      X[(size_t)(b * N_ + row) * E_ + qcol + d * 16 + l16] = acc[d][r] * inv;
  }
}

// ---------------------------------------------------------------------------
// LayerNorm over E=2048, output bf16 for final WMMA GEMM.
// ---------------------------------------------------------------------------
__global__ void __launch_bounds__(256)
layernorm_kernel(const float* __restrict__ X, const float* __restrict__ gamma,
                 const float* __restrict__ beta, unsigned short* __restrict__ XN) {
  __shared__ float s1[256], s2[256];
  const int row = blockIdx.x;
  const int tid = threadIdx.x;
  const float* x = X + (size_t)row * E_;
  float vals[8], sum = 0.f, sq = 0.f;
#pragma unroll
  for (int i = 0; i < 8; ++i) {
    float v = x[tid + i * 256];
    vals[i] = v; sum += v; sq += v * v;
  }
  s1[tid] = sum; s2[tid] = sq;
  __syncthreads();
  for (int off = 128; off > 0; off >>= 1) {
    if (tid < off) { s1[tid] += s1[tid + off]; s2[tid] += s2[tid + off]; }
    __syncthreads();
  }
  float mean = s1[0] * (1.0f / E_);
  float var  = s2[0] * (1.0f / E_) - mean * mean;
  float rstd = rsqrtf(var + 1e-5f);
#pragma unroll
  for (int i = 0; i < 8; ++i) {
    int c = tid + i * 256;
    float y = (vals[i] - mean) * rstd * gamma[c] + beta[c];
    XN[(size_t)row * E_ + c] = f32_to_bf16_rne(y);
  }
}

// ---------------------------------------------------------------------------
extern "C" void kernel_launch(void* const* d_in, const int* in_sizes, int n_in,
                              void* d_out, int out_size, void* d_ws, size_t ws_size,
                              hipStream_t stream) {
  (void)in_sizes; (void)n_in; (void)out_size; (void)ws_size;
  const float* query = (const float*)d_in[0];
  const float* key   = (const float*)d_in[1];
  const float* value = (const float*)d_in[2];
  const float* Wq    = (const float*)d_in[3];
  const float* bq    = (const float*)d_in[4];
  const float* Wk    = (const float*)d_in[5];
  const float* bk    = (const float*)d_in[6];
  const float* Wv    = (const float*)d_in[7];
  const float* bv    = (const float*)d_in[8];
  const float* gamma = (const float*)d_in[9];
  const float* beta  = (const float*)d_in[10];
  const float* Wo    = (const float*)d_in[11];
  const float* bo    = (const float*)d_in[12];
  float* out = (float*)d_out;

  // workspace layout (72 MB)
  char* ws = (char*)d_ws;
  float*          X  = (float*)ws;                                                        // 32 MB
  unsigned short* Qb = (unsigned short*)(ws + (size_t)33554432);                          // 16 MB
  unsigned short* Kb = (unsigned short*)(ws + (size_t)33554432 + 16777216);               // 4 MB
  unsigned short* Vt = (unsigned short*)(ws + (size_t)33554432 + 16777216 + 4194304);     // 4 MB
  unsigned short* XN = (unsigned short*)(ws + (size_t)33554432 + 16777216 + 2 * 4194304); // 16 MB

  const int M = B_ * N_;  // 4096
  dim3 blk(256);

  // Projections (A f32 -> bf16 on the fly); V written transposed for attention
  gemm_wmma_kernel<false, 1><<<dim3(E_ / 128, M / 128), blk, 0, stream>>>(
      query, Wq, bq, Qb, M, E_, E_);
  gemm_wmma_kernel<false, 1><<<dim3(KVE_ / 128, M / 128), blk, 0, stream>>>(
      key, Wk, bk, Kb, M, E_, KVE_);
  gemm_wmma_kernel<false, 2><<<dim3(KVE_ / 128, M / 128), blk, 0, stream>>>(
      value, Wv, bv, Vt, M, E_, KVE_);

  // Attention: one block per (b, hk, 16-row tile); 4 waves = 4 grouped q-heads
  gqa_attn_kernel<<<dim3(B_ * HK_ * (N_ / 16)), dim3(128), 0, stream>>>(
      Qb, Kb, Vt, X);

  // LayerNorm -> bf16
  layernorm_kernel<<<dim3(M), dim3(256), 0, stream>>>(X, gamma, beta, XN);

  // Output projection (A already bf16, f32 output)
  gemm_wmma_kernel<true, 0><<<dim3(E_ / 128, M / 128), blk, 0, stream>>>(
      XN, Wo, bo, out, M, E_, E_);
}
